// VarianceAdaptor_8641474199748
// MI455X (gfx1250) — matrix operand order repaired and checked by reading, not compiled, not run on hardware
//
#include <hip/hip_runtime.h>

typedef __attribute__((ext_vector_type(16))) __bf16 v16bf;
typedef __attribute__((ext_vector_type(8)))  float  v8f;

#define BB 32
#define TT 512
#define DD 384
#define K3 (3 * DD * DD)
#define MAXMEL 3584

__device__ __forceinline__ __bf16 f2bf(float x) { return (__bf16)x; }

// ---------------------------------------------------------------------------
// Weight repack: w[co][ci][k] f32 -> WMMA-fragment-ordered bf16:
//   dst[(((k*24 + mt)*12 + kb)*32 + lane)*16 + e]
// so each lane's A-fragment (v16bf) is one contiguous 32-byte load.
//   row = lane&15, hi = lane>>4
//   ci = kb*32 + hi*16 + e               (K 0..15 / 16..31)
// ---------------------------------------------------------------------------
__global__ void convw_k(const float* __restrict__ w, __bf16* __restrict__ dst) {
    int idx = blockIdx.x * blockDim.x + threadIdx.x;
    if (idx >= K3) return;
    int e    = idx & 15;
    int lane = (idx >> 4) & 31;
    int t3   = idx >> 9;          // (k*24+mt)*12 + kb
    int kb   = t3 % 12;
    int t4   = t3 / 12;           // k*24 + mt
    int mt   = t4 % 24;
    int k    = t4 / 24;
    int row  = lane & 15, hi = lane >> 4;
    int ci   = kb * 32 + hi * 16 + e;
    int co   = mt * 16 + row;
    dst[idx] = f2bf(w[co * DD * 3 + ci * 3 + k]);
}

// ---------------------------------------------------------------------------
// Zero the pad rows (row 0 and row Tc+1 per batch) of a padded bf16 act buffer
// ---------------------------------------------------------------------------
__global__ void zero_pads_k(__bf16* __restrict__ act, int Tc) {
    int idx = blockIdx.x * blockDim.x + threadIdx.x;
    if (idx >= BB * 2 * DD) return;
    int b = idx / (2 * DD);
    int r = (idx / DD) & 1;
    int d = idx % DD;
    size_t row = (size_t)b * (Tc + 2) + (r ? (Tc + 1) : 0);
    act[row * DD + d] = f2bf(0.0f);
}

// ---------------------------------------------------------------------------
// Style-adaptive norm: per (b,t) row over D, unbiased std, eps added to std.
// Writes f32 h and bf16 padded activation copy.
// ---------------------------------------------------------------------------
__global__ __launch_bounds__(128) void sadn_k(const float* __restrict__ hidden,
                                              const float* __restrict__ style,
                                              float* __restrict__ h,
                                              __bf16* __restrict__ act) {
    __shared__ float red[128];
    const int bt = blockIdx.x;
    const int b = bt >> 9, t = bt & 511;
    const int d = threadIdx.x;
    const float* x = hidden + (size_t)bt * DD;
    float v0 = x[d], v1 = x[d + 128], v2 = x[d + 256];
    red[d] = v0 + v1 + v2;
    __syncthreads();
    for (int o = 64; o > 0; o >>= 1) { if (d < o) red[d] += red[d + o]; __syncthreads(); }
    float mean = red[0] * (1.0f / DD);
    __syncthreads();
    float d0 = v0 - mean, d1 = v1 - mean, d2 = v2 - mean;
    red[d] = d0 * d0 + d1 * d1 + d2 * d2;
    __syncthreads();
    for (int o = 64; o > 0; o >>= 1) { if (d < o) red[d] += red[d + o]; __syncthreads(); }
    float var = red[0] * (1.0f / (DD - 1));           // unbiased (ddof=1)
    float inv = 1.0f / (sqrtf(var) + 1e-5f);          // eps added to std
    const float* st = style + (size_t)b * DD;
    float* ho = h + (size_t)bt * DD;
    __bf16* ao = act + ((size_t)b * (TT + 2) + t + 1) * DD;
    float y0 = d0 * inv + st[d];
    float y1 = d1 * inv + st[d + 128];
    float y2 = d2 * inv + st[d + 256];
    ho[d] = y0;       ao[d] = f2bf(y0);
    ho[d + 128] = y1; ao[d + 128] = f2bf(y1);
    ho[d + 256] = y2; ao[d + 256] = f2bf(y2);
}

// ---------------------------------------------------------------------------
// Conv1d(D->D, k=3) as bf16 WMMA GEMM with fp32 accumulation + bias.
// act: padded [B, Tc+2, D] bf16 ; Wsw: fragment-ordered [3][24][12][512] bf16
// out: [B, Tc, D] f32.  Grid = (ceil(Tc/64), 3, B); block = 8 waves (4M x 2N);
// wave tile 32x32; K = 3 taps x 384 channels.
// ---------------------------------------------------------------------------
__global__ __launch_bounds__(256) void conv_wmma_k(const __bf16* __restrict__ act,
                                                   const __bf16* __restrict__ Wsw,
                                                   const float* __restrict__ bias,
                                                   float* __restrict__ out, int Tc) {
    const int tid = threadIdx.x;
    const int lane = tid & 31;
    const int wave = tid >> 5;
    const int waveM = wave & 3;
    const int waveN = wave >> 2;
    const int b = blockIdx.z;
    const int coW = blockIdx.y * 128 + waveM * 32;
    const int mt0 = blockIdx.y * 8 + waveM * 2;      // M-tile index (of 16 rows)
    const int tW = blockIdx.x * 64 + waveN * 32;
    const int row = lane & 15;
    const int hi = lane >> 4;

    // B-fragment per-column base pointers (one lane = one output column).
    const __bf16* colBase[2];
    int tcol[2];
#pragma unroll
    for (int j = 0; j < 2; ++j) {
        int t = tW + j * 16 + row;
        tcol[j] = t;
        int tc = (t < Tc) ? t : (Tc - 1);
        colBase[j] = act + ((size_t)b * (Tc + 2) + tc) * DD + hi * 16;
    }

    v8f acc[2][2];
#pragma unroll
    for (int i = 0; i < 2; ++i)
#pragma unroll
        for (int j = 0; j < 2; ++j)
#pragma unroll
            for (int e = 0; e < 8; ++e) acc[i][j][e] = 0.0f;

    for (int k = 0; k < 3; ++k) {
        // fragment-ordered weight base for this tap: [k][mt][kb][lane][16]
        const __bf16* wBase0 = Wsw + ((size_t)((k * 24 + mt0) * 12) * 32 + lane) * 16;
        const __bf16* wBase1 = wBase0 + (size_t)12 * 32 * 16;   // next M-tile
        const size_t aOff = (size_t)k * DD;                      // tap shift = 1 padded row
#pragma unroll
        for (int kb = 0; kb < 12; ++kb) {
            v16bf a0 = *(const v16bf*)(wBase0 + (size_t)kb * 512);
            v16bf a1 = *(const v16bf*)(wBase1 + (size_t)kb * 512);
            v16bf b0 = *(const v16bf*)(colBase[0] + aOff + kb * 32);
            v16bf b1 = *(const v16bf*)(colBase[1] + aOff + kb * 32);
            acc[0][0] = __builtin_amdgcn_wmma_f32_16x16x32_bf16(
                false, a0, false, b0, (short)0, acc[0][0], false, false);
            acc[0][1] = __builtin_amdgcn_wmma_f32_16x16x32_bf16(
                false, a0, false, b1, (short)0, acc[0][1], false, false);
            acc[1][0] = __builtin_amdgcn_wmma_f32_16x16x32_bf16(
                false, a1, false, b0, (short)0, acc[1][0], false, false);
            acc[1][1] = __builtin_amdgcn_wmma_f32_16x16x32_bf16(
                false, a1, false, b1, (short)0, acc[1][1], false, false);
        }
    }

#pragma unroll
    for (int i = 0; i < 2; ++i) {
        int co0 = coW + i * 16 + hi * 8;
#pragma unroll
        for (int j = 0; j < 2; ++j) {
            if (tcol[j] < Tc) {
                float* op = out + ((size_t)b * Tc + tcol[j]) * DD + co0;
#pragma unroll
                for (int r = 0; r < 8; ++r) op[r] = acc[i][j][r] + bias[co0 + r];
            }
        }
    }
}

// ---------------------------------------------------------------------------
// ReLU + LayerNorm(channels) + mask -> bf16 padded activation buffer
// ---------------------------------------------------------------------------
__global__ __launch_bounds__(128) void ln_mask_k(const float* __restrict__ in,
                                                 const float* __restrict__ g,
                                                 const float* __restrict__ be,
                                                 const float* __restrict__ mask,
                                                 __bf16* __restrict__ actOut, int Tc) {
    __shared__ float red[128];
    const int bt = blockIdx.x;
    const int b = bt / Tc, t = bt % Tc;
    const int d = threadIdx.x;
    const float* x = in + (size_t)bt * DD;
    float v0 = fmaxf(x[d], 0.f), v1 = fmaxf(x[d + 128], 0.f), v2 = fmaxf(x[d + 256], 0.f);
    red[d] = v0 + v1 + v2;
    __syncthreads();
    for (int o = 64; o > 0; o >>= 1) { if (d < o) red[d] += red[d + o]; __syncthreads(); }
    float mean = red[0] * (1.0f / DD);
    __syncthreads();
    float d0 = v0 - mean, d1 = v1 - mean, d2 = v2 - mean;
    red[d] = d0 * d0 + d1 * d1 + d2 * d2;
    __syncthreads();
    for (int o = 64; o > 0; o >>= 1) { if (d < o) red[d] += red[d + o]; __syncthreads(); }
    float rstd = rsqrtf(red[0] * (1.0f / DD) + 1e-5f);
    float mk = mask[bt];
    __bf16* o = actOut + ((size_t)b * (Tc + 2) + t + 1) * DD;
    o[d]       = f2bf((d0 * rstd * g[d] + be[d]) * mk);
    o[d + 128] = f2bf((d1 * rstd * g[d + 128] + be[d + 128]) * mk);
    o[d + 256] = f2bf((d2 * rstd * g[d + 256] + be[d + 256]) * mk);
}

// ---------------------------------------------------------------------------
// ReLU + LN + mask + dot(lw) + lb -> pred[b,t]
// ---------------------------------------------------------------------------
__global__ __launch_bounds__(128) void final_pred_k(const float* __restrict__ in,
                                                    const float* __restrict__ g,
                                                    const float* __restrict__ be,
                                                    const float* __restrict__ mask,
                                                    const float* __restrict__ lw,
                                                    const float* __restrict__ lb,
                                                    float* __restrict__ pred, int Tc) {
    __shared__ float red[128];
    const int bt = blockIdx.x;
    const int d = threadIdx.x;
    const float* x = in + (size_t)bt * DD;
    float v0 = fmaxf(x[d], 0.f), v1 = fmaxf(x[d + 128], 0.f), v2 = fmaxf(x[d + 256], 0.f);
    red[d] = v0 + v1 + v2;
    __syncthreads();
    for (int o = 64; o > 0; o >>= 1) { if (d < o) red[d] += red[d + o]; __syncthreads(); }
    float mean = red[0] * (1.0f / DD);
    __syncthreads();
    float d0 = v0 - mean, d1 = v1 - mean, d2 = v2 - mean;
    red[d] = d0 * d0 + d1 * d1 + d2 * d2;
    __syncthreads();
    for (int o = 64; o > 0; o >>= 1) { if (d < o) red[d] += red[d + o]; __syncthreads(); }
    float rstd = rsqrtf(red[0] * (1.0f / DD) + 1e-5f);
    __syncthreads();
    red[d] = (d0 * rstd * g[d] + be[d]) * lw[d]
           + (d1 * rstd * g[d + 128] + be[d + 128]) * lw[d + 128]
           + (d2 * rstd * g[d + 256] + be[d + 256]) * lw[d + 256];
    __syncthreads();
    for (int o = 64; o > 0; o >>= 1) { if (d < o) red[d] += red[d + o]; __syncthreads(); }
    if (d == 0) pred[bt] = mask[bt] * red[0] + lb[0];
}

// ---------------------------------------------------------------------------
// Masked inclusive cumsum of durations per batch row
// ---------------------------------------------------------------------------
__global__ void cumsum_k(const int* __restrict__ dur, const float* __restrict__ mask,
                         int* __restrict__ cum) {
    int b = threadIdx.x;
    if (b >= BB) return;
    int s = 0;
    for (int t = 0; t < TT; ++t) {
        int i = b * TT + t;
        s += (mask[i] > 0.f) ? dur[i] : 0;
        cum[i] = s;
    }
}

// ---------------------------------------------------------------------------
// Length regulate: binary search cum for each mel slot; writes exp_h (f32,
// into d_out), bf16 padded act copy, and mel_mask.
// ---------------------------------------------------------------------------
__global__ __launch_bounds__(128) void length_reg_k(const float* __restrict__ h,
                                                    const int* __restrict__ cum,
                                                    float* __restrict__ expOut,
                                                    __bf16* __restrict__ act,
                                                    float* __restrict__ melmask, int mel) {
    __shared__ int sidx;
    __shared__ float sval;
    const int bm = blockIdx.x;
    const int b = bm / mel, m = bm % mel;
    if (threadIdx.x == 0) {
        const int* c = cum + b * TT;
        int lo = 0, hiq = TT;
        while (lo < hiq) { int mid = (lo + hiq) >> 1; if (c[mid] <= m) lo = mid + 1; else hiq = mid; }
        sidx = (lo < TT - 1) ? lo : (TT - 1);
        float valid = (m < c[TT - 1]) ? 1.f : 0.f;
        sval = valid;
        melmask[bm] = valid;
    }
    __syncthreads();
    const int idx = sidx;
    const float valid = sval;
    const float* src = h + ((size_t)b * TT + idx) * DD;
    float* dst = expOut + (size_t)bm * DD;
    __bf16* ad = act + ((size_t)b * (mel + 2) + m + 1) * DD;
    for (int d = threadIdx.x; d < DD; d += 128) {
        float v = src[d] * valid;
        dst[d] = v;
        ad[d] = f2bf(v);
    }
}

// ---------------------------------------------------------------------------
// 1->D conv1d (k=9, pad 4) embedding, added in place to exp_h; optionally
// refreshes the bf16 activation copy for the next predictor.
// ---------------------------------------------------------------------------
__global__ void emb_conv_k(const float* __restrict__ w, const float* __restrict__ bias,
                           const float* __restrict__ pred, float* __restrict__ expOut,
                           __bf16* __restrict__ act, int mel, int writeAct) {
    size_t idx = (size_t)blockIdx.x * blockDim.x + threadIdx.x;
    size_t tot = (size_t)BB * mel * DD;
    if (idx >= tot) return;
    int co = (int)(idx % DD);
    size_t bm = idx / DD;
    int m = (int)(bm % mel);
    int b = (int)(bm / mel);
    const float* p = pred + (size_t)b * mel;
    float s = bias[co];
#pragma unroll
    for (int k = 0; k < 9; ++k) {
        int mp = m + k - 4;
        if (mp >= 0 && mp < mel) s += w[co * 9 + k] * p[mp];
    }
    float v = expOut[idx] + s;
    expOut[idx] = v;
    if (writeAct) act[((size_t)b * (mel + 2) + m + 1) * DD + co] = f2bf(v);
}

// ---------------------------------------------------------------------------
extern "C" void kernel_launch(void* const* d_in, const int* in_sizes, int n_in,
                              void* d_out, int out_size, void* d_ws, size_t ws_size,
                              hipStream_t stream) {
    (void)in_sizes; (void)n_in; (void)ws_size;
    auto F = [&](int i) { return (const float*)d_in[i]; };
    const float* hidden = F(0);
    const float* mask = F(1);
    const float* style = F(2);
    const int* dur = (const int*)d_in[3];
    // dp params at 4..13, pp at 14..23, ep at 24..33 (w1,b1,g1,be1,w2,b2,g2,be2,lw,lb)
    const float* pe_w = F(34); const float* pe_b = F(35);
    const float* ee_w = F(36); const float* ee_b = F(37);

    // mel_len recovered from output size: out = B*mel*D + B*T + 2*B*mel
    int mel = (int)(((size_t)out_size - (size_t)BB * TT) / ((size_t)BB * (DD + 2)));
    if (mel < 1) mel = 1;
    if (mel > MAXMEL) mel = MAXMEL;

    auto al = [](size_t x) { return (x + 255) & ~(size_t)255; };
    char* base = (char*)d_ws;
    size_t off = 0;
    float* h = (float*)(base + off);        off += al((size_t)BB * TT * DD * 4);
    int* cum = (int*)(base + off);          off += al((size_t)BB * TT * 4);
    float* melmask = (float*)(base + off);  off += al((size_t)BB * MAXMEL * 4);
    __bf16* actA = (__bf16*)(base + off);   off += al((size_t)BB * (MAXMEL + 2) * DD * 2);
    __bf16* actB = (__bf16*)(base + off);   off += al((size_t)BB * (MAXMEL + 2) * DD * 2);
    float* scratch = (float*)(base + off);  off += al((size_t)BB * MAXMEL * DD * 4);
    __bf16* Wbf = (__bf16*)(base + off);    off += al((size_t)6 * K3 * 2);

    const int wgrid = (K3 + 255) / 256;
    convw_k<<<wgrid, 256, 0, stream>>>(F(4),  Wbf + 0 * (size_t)K3);   // dp.w1
    convw_k<<<wgrid, 256, 0, stream>>>(F(8),  Wbf + 1 * (size_t)K3);   // dp.w2
    convw_k<<<wgrid, 256, 0, stream>>>(F(14), Wbf + 2 * (size_t)K3);   // pp.w1
    convw_k<<<wgrid, 256, 0, stream>>>(F(18), Wbf + 3 * (size_t)K3);   // pp.w2
    convw_k<<<wgrid, 256, 0, stream>>>(F(24), Wbf + 4 * (size_t)K3);   // ep.w1
    convw_k<<<wgrid, 256, 0, stream>>>(F(28), Wbf + 5 * (size_t)K3);   // ep.w2

    float* out = (float*)d_out;
    float* expOut = out;
    float* predDur = out + (size_t)BB * mel * DD;
    float* predPitch = predDur + (size_t)BB * TT;
    float* predEnergy = predPitch + (size_t)BB * mel;

    const int padGrid = (BB * 2 * DD + 255) / 256;

    auto runVP = [&](const __bf16* actIn, __bf16* actMid, int pbase,
                     const __bf16* Wc, const float* mk, int Tc, float* predOut) {
        int Ntot = BB * Tc;
        dim3 g((Tc + 63) / 64, 3, BB);
        conv_wmma_k<<<g, 256, 0, stream>>>(actIn, Wc, F(pbase + 1), scratch, Tc);
        zero_pads_k<<<padGrid, 256, 0, stream>>>(actMid, Tc);
        ln_mask_k<<<Ntot, 128, 0, stream>>>(scratch, F(pbase + 2), F(pbase + 3), mk, actMid, Tc);
        conv_wmma_k<<<g, 256, 0, stream>>>(actMid, Wc + (size_t)K3, F(pbase + 5), scratch, Tc);
        final_pred_k<<<Ntot, 128, 0, stream>>>(scratch, F(pbase + 6), F(pbase + 7), mk,
                                               F(pbase + 8), F(pbase + 9), predOut, Tc);
    };

    // style-adaptive norm + duration predictor (N = B*T)
    zero_pads_k<<<padGrid, 256, 0, stream>>>(actA, TT);
    sadn_k<<<BB * TT, 128, 0, stream>>>(hidden, style, h, actA);
    runVP(actA, actB, 4, Wbf, mask, TT, predDur);

    // length regulate (N = B*mel)
    cumsum_k<<<1, 32, 0, stream>>>(dur, mask, cum);
    zero_pads_k<<<padGrid, 256, 0, stream>>>(actA, mel);
    length_reg_k<<<BB * mel, 128, 0, stream>>>(h, cum, expOut, actA, melmask, mel);

    // pitch predictor + embedding
    runVP(actA, actB, 14, Wbf + 2 * (size_t)K3, melmask, mel, predPitch);
    size_t embTot = (size_t)BB * mel * DD;
    int embGrid = (int)((embTot + 255) / 256);
    emb_conv_k<<<embGrid, 256, 0, stream>>>(pe_w, pe_b, predPitch, expOut, actA, mel, 1);

    // energy predictor + embedding
    runVP(actA, actB, 24, Wbf + 4 * (size_t)K3, melmask, mel, predEnergy);
    emb_conv_k<<<embGrid, 256, 0, stream>>>(ee_w, ee_b, predEnergy, expOut, actA, mel, 0);
}
